// acLSTM_43937515438236
// MI455X (gfx1250) — matrix operand order repaired and checked by reading, not compiled, not run on hardware
//
#include <hip/hip_runtime.h>
#include <stdint.h>

#define BATCH   64
#define SEQLEN  160
#define NIN     171
#define HID     1024
#define LDXP    192      // padded input width (multiple of 32)
#define NOUT    171
#define LDRAW   176      // padded decoder width
#define NWG     32
#define NTHR    256      // 8 waves per WG -> 256 waves total
#define HS      (BATCH*HID)

typedef __attribute__((ext_vector_type(16))) __bf16 v16bf;
typedef __attribute__((ext_vector_type(8)))  float  v8f;

// ---------- helpers ----------

__device__ __forceinline__ unsigned short f2bf(float x) {
  unsigned u = __float_as_uint(x);
  u += 0x7FFFu + ((u >> 16) & 1u);
  return (unsigned short)(u >> 16);
}

__device__ __forceinline__ unsigned pack2bf(float a, float b) {
  unsigned ua = __float_as_uint(a); ua += 0x7FFFu + ((ua >> 16) & 1u);
  unsigned ub = __float_as_uint(b); ub += 0x7FFFu + ((ub >> 16) & 1u);
  return (ua >> 16) | (ub & 0xFFFF0000u);
}

__device__ __forceinline__ float sigf(float x) {
  return 1.0f / (1.0f + __expf(-x));
}
__device__ __forceinline__ float tanh_(float x) {
  float e = __expf(2.0f * x);
  return 1.0f - 2.0f / (e + 1.0f);
}

// A fragment (16x32 bf16) per ISA layout: lane holds M=lane&15,
// K pairs: VGPR v<4 -> k = hi*8 + 2v ; v>=4 -> k = 16 + hi*8 + 2(v-4).
// With bf16 row-major activations this is exactly two b128 loads per lane.
__device__ __forceinline__ v16bf load_a_frag(const unsigned short* __restrict__ X, int ld,
                                             int row, int kbase, int lane) {
  union { v16bf v; uint4 q[2]; } a;
  const unsigned short* p = X + (size_t)row * ld + kbase + ((lane >> 4) & 1) * 8;
  a.q[0] = *(const uint4*)(p);        // k offsets 0..7   -> VGPR 0..3
  a.q[1] = *(const uint4*)(p + 16);   // k offsets 16..23 -> VGPR 4..7
  return a.v;
}

// B fragment pre-swizzled in ws: layout [frag][lane][8] u32 -> two b128 loads
__device__ __forceinline__ v16bf load_b_frag(const unsigned* __restrict__ Wp,
                                             size_t frag, int lane) {
  union { v16bf v; uint4 q[2]; } b;
  const uint4* p = (const uint4*)(Wp + frag * 256 + (size_t)lane * 8);
  b.q[0] = p[0]; b.q[1] = p[1];
  return b.v;
}

__device__ __forceinline__ v8f wmma_bf16(v16bf a, v16bf b, v8f c) {
  return __builtin_amdgcn_wmma_f32_16x16x32_bf16(false, a, false, b, (short)0, c,
                                                 false, false);
}

// device-wide sense-reversing barrier (all NWG blocks co-resident)
__device__ __forceinline__ void gbar(unsigned* bar) {
  __threadfence();          // release stores to device scope (L2)
  __syncthreads();
  if (threadIdx.x == 0) {
    unsigned gen = __hip_atomic_load(bar + 1, __ATOMIC_RELAXED, __HIP_MEMORY_SCOPE_AGENT);
    unsigned arrived = __hip_atomic_fetch_add(bar, 1u, __ATOMIC_ACQ_REL, __HIP_MEMORY_SCOPE_AGENT);
    if (arrived == NWG - 1) {
      __hip_atomic_store(bar, 0u, __ATOMIC_RELAXED, __HIP_MEMORY_SCOPE_AGENT);
      __hip_atomic_fetch_add(bar + 1, 1u, __ATOMIC_RELEASE, __HIP_MEMORY_SCOPE_AGENT);
    } else {
      while (__hip_atomic_load(bar + 1, __ATOMIC_ACQUIRE, __HIP_MEMORY_SCOPE_AGENT) == gen) {
        __builtin_amdgcn_s_sleep(8);
      }
    }
  }
  __syncthreads();
  __threadfence();          // acquire: invalidate stale near caches
}

// ---------- prep kernels ----------

// Pre-swizzle weight W[N][K] (fp32, row-major) into bf16 B-fragment layout.
// dst element idx -> ntile, ktile, lane, v ; B(k,n)=W[n][k]
__global__ void pack_weight(const float* __restrict__ W, unsigned* __restrict__ dst,
                            int N, int K, int nTiles, int kTiles) {
  size_t idx = (size_t)blockIdx.x * 256 + threadIdx.x;
  size_t total = (size_t)nTiles * kTiles * 256;
  if (idx >= total) return;
  int nt  = (int)(idx / ((size_t)kTiles * 256));
  int rem = (int)(idx % ((size_t)kTiles * 256));
  int kt = rem >> 8;
  int r2 = rem & 255;
  int l = r2 >> 3, v = r2 & 7;
  int hi = l >> 4;
  int n = nt * 16 + (l & 15);
  int k = kt * 32 + hi * 8 + 2 * v + ((v >> 2) << 3);
  float a = 0.f, b = 0.f;
  if (n < N) {
    if (k < K)     a = W[(size_t)n * K + k];
    if (k + 1 < K) b = W[(size_t)n * K + k + 1];
  }
  dst[idx] = pack2bf(a, b);
}

__global__ void init_kernel(const float* __restrict__ bih1, const float* __restrict__ bhh1,
                            const float* __restrict__ bih2, const float* __restrict__ bhh2,
                            const float* __restrict__ bih3, const float* __restrict__ bhh3,
                            const float* __restrict__ bdec,
                            const float* __restrict__ xseq,
                            const int* __restrict__ cnum, const int* __restrict__ gnum,
                            float* __restrict__ bias1, float* __restrict__ bias2,
                            float* __restrict__ bias3, float* __restrict__ biasd,
                            unsigned short* __restrict__ hbase,
                            unsigned short* __restrict__ xbuf,
                            unsigned* __restrict__ bar) {
  int idx = blockIdx.x * 256 + threadIdx.x;
  if (idx < 4096) {
    bias1[idx] = bih1[idx] + bhh1[idx];
    bias2[idx] = bih2[idx] + bhh2[idx];
    bias3[idx] = bih3[idx] + bhh3[idx];
  }
  if (idx < LDRAW) biasd[idx] = (idx < NOUT) ? bdec[idx] : 0.f;
  if (idx < 2) bar[idx] = 0u;
  for (size_t i = idx; i < (size_t)6 * HS; i += (size_t)gridDim.x * 256) hbase[i] = 0;
  if (idx < BATCH * LDXP) {
    int b = idx / LDXP, d = idx % LDXP;
    float v = 0.f;
    if (d < NIN) {
      int gt = *gnum; int per = gt + *cnum; if (per < 1) per = 1;
      if ((0 % per) < gt) v = xseq[(size_t)b * SEQLEN * NIN + d];
    }
    xbuf[idx] = f2bf(v);
  }
}

// ---------- main persistent kernel ----------

// One wave owns (mtile, nslice): computes i,f,g,o gate tiles for its 16-col
// slice of h, keeps cell state c in VGPRs across all 160 steps.
__device__ __forceinline__ void lstm_layer(const unsigned short* __restrict__ Xa, int ldx, int kTa,
                                           const unsigned* __restrict__ Wih,
                                           const unsigned short* __restrict__ Ha,
                                           const unsigned* __restrict__ Whh,
                                           const float* __restrict__ bias,
                                           unsigned short* __restrict__ Hout,
                                           v8f& cst, int mtile, int nsl, int lane) {
  v8f acc[4];
#pragma unroll
  for (int g = 0; g < 4; ++g)
#pragma unroll
    for (int i = 0; i < 8; ++i) acc[g][i] = 0.f;

  const int arow = mtile * 16 + (lane & 15);

#pragma unroll 2
  for (int kt = 0; kt < kTa; ++kt) {           // input-hidden part
    v16bf a = load_a_frag(Xa, ldx, arow, kt * 32, lane);
#pragma unroll
    for (int g = 0; g < 4; ++g) {
      v16bf b = load_b_frag(Wih, (size_t)(g * 64 + nsl) * kTa + kt, lane);
      acc[g] = wmma_bf16(a, b, acc[g]);
    }
  }
#pragma unroll 2
  for (int kt = 0; kt < 32; ++kt) {            // hidden-hidden part (K=1024)
    v16bf a = load_a_frag(Ha, HID, arow, kt * 32, lane);
#pragma unroll
    for (int g = 0; g < 4; ++g) {
      v16bf b = load_b_frag(Whh, (size_t)(g * 64 + nsl) * 32 + kt, lane);
      acc[g] = wmma_bf16(a, b, acc[g]);
    }
  }

  const int col = lane & 15, hi = (lane >> 4) & 1;
  const int n = nsl * 16 + col;
  const float b0 = bias[n],           b1 = bias[HID + n];
  const float b2 = bias[2 * HID + n], b3 = bias[3 * HID + n];
#pragma unroll
  for (int v = 0; v < 8; ++v) {
    float iv = sigf(acc[0][v] + b0);
    float fv = sigf(acc[1][v] + b1);
    float gv = tanh_(acc[2][v] + b2);
    float ov = sigf(acc[3][v] + b3);
    float cn = fv * cst[v] + iv * gv;
    cst[v] = cn;
    float hv = ov * tanh_(cn);
    int m = mtile * 16 + v + 8 * hi;
    Hout[(size_t)m * HID + n] = f2bf(hv);      // bf16 once per value
  }
}

__global__ __launch_bounds__(NTHR) void aclstm_persistent(
    const float* __restrict__ xseq, const int* __restrict__ cnum, const int* __restrict__ gnum,
    const unsigned* __restrict__ Wih1, const unsigned* __restrict__ Whh1,
    const unsigned* __restrict__ Wih2, const unsigned* __restrict__ Whh2,
    const unsigned* __restrict__ Wih3, const unsigned* __restrict__ Whh3,
    const unsigned* __restrict__ Wdec,
    const float* __restrict__ bias1, const float* __restrict__ bias2,
    const float* __restrict__ bias3, const float* __restrict__ biasd,
    unsigned short* __restrict__ hbase, unsigned short* __restrict__ xbuf,
    float* __restrict__ raw, unsigned* __restrict__ bar, float* __restrict__ out) {
  const int lane  = threadIdx.x & 31;
  const int wid   = (blockIdx.x << 3) + (threadIdx.x >> 5);   // 0..255
  const int mtile = wid & 3;
  const int nsl   = wid >> 2;                                  // 0..63
  const int tid   = blockIdx.x * NTHR + threadIdx.x;

  v8f c0, c1, c2;
#pragma unroll
  for (int i = 0; i < 8; ++i) { c0[i] = 0.f; c1[i] = 0.f; c2[i] = 0.f; }

  const int gt = *gnum;
  int per = gt + *cnum; if (per < 1) per = 1;

  for (int t = 0; t < SEQLEN; ++t) {
    const int pr = t & 1, pw = pr ^ 1;
    const unsigned short* h0r = hbase + (size_t)(0 + pr) * HS;
    unsigned short*       h0w = hbase + (size_t)(0 + pw) * HS;
    const unsigned short* h1r = hbase + (size_t)(2 + pr) * HS;
    unsigned short*       h1w = hbase + (size_t)(2 + pw) * HS;
    const unsigned short* h2r = hbase + (size_t)(4 + pr) * HS;
    unsigned short*       h2w = hbase + (size_t)(4 + pw) * HS;

    lstm_layer(xbuf, LDXP, 6, Wih1, h0r, Whh1, bias1, h0w, c0, mtile, nsl, lane);
    gbar(bar);
    lstm_layer(h0w, HID, 32, Wih2, h1r, Whh2, bias2, h1w, c1, mtile, nsl, lane);
    gbar(bar);
    lstm_layer(h1w, HID, 32, Wih3, h2r, Whh3, bias3, h2w, c2, mtile, nsl, lane);
    gbar(bar);

    // decoder GEMM: 4 mtiles x 11 ntiles = 44 wave-jobs
    if (wid < 44) {
      int dmt = wid & 3, dnt = wid >> 2;
      v8f acc;
#pragma unroll
      for (int i = 0; i < 8; ++i) acc[i] = 0.f;
      const int arow = dmt * 16 + (lane & 15);
#pragma unroll 2
      for (int kt = 0; kt < 32; ++kt) {
        v16bf a = load_a_frag(h2w, HID, arow, kt * 32, lane);
        v16bf b = load_b_frag(Wdec, (size_t)dnt * 32 + kt, lane);
        acc = wmma_bf16(a, b, acc);
      }
      const int col = lane & 15, hi = (lane >> 4) & 1;
      const int n = dnt * 16 + col;
      const float bb = biasd[n];
#pragma unroll
      for (int v = 0; v < 8; ++v) {
        int m = dmt * 16 + v + 8 * hi;
        raw[(size_t)m * LDRAW + n] = acc[v] + bb;
      }
    }
    gbar(bar);

    // pointwise: hip copy + quaternion normalize, write out, build next x (bf16)
    if (tid < BATCH * 43) {
      int b = tid / 43, s = tid % 43;
      const float* r = raw + (size_t)b * LDRAW;
      float* ob = out + (size_t)b * SEQLEN * NOUT + (size_t)t * NOUT;
      int nt2 = t + 1;
      int cond = (nt2 < SEQLEN) ? (((nt2 % per) < gt) ? 1 : 0) : 0;
      const float* xr = xseq + (size_t)b * SEQLEN * NIN + (size_t)nt2 * NIN;
      unsigned short* xb = xbuf + (size_t)b * LDXP;
      if (s == 0) {
#pragma unroll
        for (int d = 0; d < 3; ++d) {
          float vv = r[d];
          ob[d] = vv;
          if (nt2 < SEQLEN) xb[d] = f2bf(cond ? xr[d] : vv);
        }
      } else {
        int d0 = 3 + 4 * (s - 1);
        float q0 = r[d0], q1 = r[d0 + 1], q2 = r[d0 + 2], q3 = r[d0 + 3];
        float den = fmaxf(sqrtf(q0 * q0 + q1 * q1 + q2 * q2 + q3 * q3), 1e-6f);
        float o0 = q0 / den, o1 = q1 / den, o2 = q2 / den, o3 = q3 / den;
        ob[d0] = o0; ob[d0 + 1] = o1; ob[d0 + 2] = o2; ob[d0 + 3] = o3;
        if (nt2 < SEQLEN) {
          xb[d0]     = f2bf(cond ? xr[d0]     : o0);
          xb[d0 + 1] = f2bf(cond ? xr[d0 + 1] : o1);
          xb[d0 + 2] = f2bf(cond ? xr[d0 + 2] : o2);
          xb[d0 + 3] = f2bf(cond ? xr[d0 + 3] : o3);
        }
      }
    }
    gbar(bar);
  }
}

// ---------- host ----------

extern "C" void kernel_launch(void* const* d_in, const int* in_sizes, int n_in,
                              void* d_out, int out_size, void* d_ws, size_t ws_size,
                              hipStream_t stream) {
  (void)in_sizes; (void)n_in; (void)out_size; (void)ws_size;
  const float* xseq = (const float*)d_in[0];
  const int*   cn   = (const int*)d_in[1];
  const int*   gn   = (const int*)d_in[2];
  const float* Wih1 = (const float*)d_in[3];
  const float* Whh1 = (const float*)d_in[4];
  const float* bih1 = (const float*)d_in[5];
  const float* bhh1 = (const float*)d_in[6];
  const float* Wih2 = (const float*)d_in[7];
  const float* Whh2 = (const float*)d_in[8];
  const float* bih2 = (const float*)d_in[9];
  const float* bhh2 = (const float*)d_in[10];
  const float* Wih3 = (const float*)d_in[11];
  const float* Whh3 = (const float*)d_in[12];
  const float* bih3 = (const float*)d_in[13];
  const float* bhh3 = (const float*)d_in[14];
  const float* Wdec = (const float*)d_in[15];
  const float* bdec = (const float*)d_in[16];

  char* ws = (char*)d_ws;
  size_t off = 0;
  auto alloc = [&](size_t bytes) -> void* {
    void* p = ws + off;
    off = (off + bytes + 255) & ~(size_t)255;
    return p;
  };
  const size_t szSmall = (size_t)256 * 6  * 256 * 4;   // Wih1 packed
  const size_t szBig   = (size_t)256 * 32 * 256 * 4;   // H->4H packed
  const size_t szDec   = (size_t)11  * 32 * 256 * 4;   // Wdec packed
  unsigned* Wih1p = (unsigned*)alloc(szSmall);
  unsigned* Whh1p = (unsigned*)alloc(szBig);
  unsigned* Wih2p = (unsigned*)alloc(szBig);
  unsigned* Whh2p = (unsigned*)alloc(szBig);
  unsigned* Wih3p = (unsigned*)alloc(szBig);
  unsigned* Whh3p = (unsigned*)alloc(szBig);
  unsigned* Wdecp = (unsigned*)alloc(szDec);
  float* bias1 = (float*)alloc(4096 * 4);
  float* bias2 = (float*)alloc(4096 * 4);
  float* bias3 = (float*)alloc(4096 * 4);
  float* biasd = (float*)alloc(LDRAW * 4);
  unsigned short* hbase = (unsigned short*)alloc((size_t)6 * HS * 2);
  unsigned short* xbufp = (unsigned short*)alloc((size_t)BATCH * LDXP * 2);
  float* rawp  = (float*)alloc((size_t)BATCH * LDRAW * 4);
  unsigned* bar = (unsigned*)alloc(256);

  // 1) pre-swizzle all weights to bf16 B-fragment layout (one-time, HBM -> L2 resident)
  pack_weight<<<(unsigned)((szSmall / 4 + 255) / 256), 256, 0, stream>>>(Wih1, Wih1p, 4096, 171, 256, 6);
  pack_weight<<<(unsigned)((szBig  / 4 + 255) / 256), 256, 0, stream>>>(Whh1, Whh1p, 4096, 1024, 256, 32);
  pack_weight<<<(unsigned)((szBig  / 4 + 255) / 256), 256, 0, stream>>>(Wih2, Wih2p, 4096, 1024, 256, 32);
  pack_weight<<<(unsigned)((szBig  / 4 + 255) / 256), 256, 0, stream>>>(Whh2, Whh2p, 4096, 1024, 256, 32);
  pack_weight<<<(unsigned)((szBig  / 4 + 255) / 256), 256, 0, stream>>>(Wih3, Wih3p, 4096, 1024, 256, 32);
  pack_weight<<<(unsigned)((szBig  / 4 + 255) / 256), 256, 0, stream>>>(Whh3, Whh3p, 4096, 1024, 256, 32);
  pack_weight<<<(unsigned)((szDec  / 4 + 255) / 256), 256, 0, stream>>>(Wdec, Wdecp, 171, 1024, 11, 32);

  // 2) state / bias init (re-done every call: no cross-call state)
  init_kernel<<<1536, 256, 0, stream>>>(bih1, bhh1, bih2, bhh2, bih3, bhh3, bdec,
                                        xseq, cn, gn, bias1, bias2, bias3, biasd,
                                        hbase, xbufp, bar);

  // 3) single persistent kernel: 160 timesteps, WMMA bf16, device barriers
  aclstm_persistent<<<NWG, NTHR, 0, stream>>>(
      xseq, cn, gn, Wih1p, Whh1p, Wih2p, Whh2p, Wih3p, Whh3p, Wdecp,
      bias1, bias2, bias3, biasd, hbase, xbufp, rawp, bar, (float*)d_out);
}
